// SelfAttention_2044404433362
// MI455X (gfx1250) — matrix operand order repaired and checked by reading, not compiled, not run on hardware
//
#include <hip/hip_runtime.h>
#include <hip/hip_bf16.h>

typedef __attribute__((ext_vector_type(16))) _Float16 v16h;
typedef __attribute__((ext_vector_type(8)))  _Float16 v8h;
typedef __attribute__((ext_vector_type(8)))  float    v8f;
typedef __attribute__((ext_vector_type(4)))  int      v4i;

union V16 { v16h v; v8h h[2]; };

#define N_EMBD   1024
#define N_HEAD   16
#define HEAD_DIM 64
#define BATCH    2
#define SEQ      2048
#define ROWS_BT  (BATCH * SEQ)       // 4096 rows
#define C3       (3 * N_EMBD)        // 3072

// ---------------- workspace layout (bytes) ----------------
#define OFF_XH     ((size_t)0)                 // f16 x        [BT, C]      8 MB
#define OFF_WQKVT  ((size_t)8  << 20)          // f16 Wqkv^T   [3C, C]      6 MB
#define OFF_WPROJT ((size_t)14 << 20)          // f16 Wproj^T  [C, C]       2 MB
#define OFF_QKV    ((size_t)16 << 20)          // f16 qkv      [BT, 3C]    24 MB
#define OFF_QH     ((size_t)40 << 20)          // f16 q(rope)  [BH, T, D]   8 MB
#define OFF_KH     ((size_t)48 << 20)          // f16 k(rope)  [BH, T, D]   8 MB
#define OFF_VT     ((size_t)56 << 20)          // f16 v^T      [BH, D, T]   8 MB
#define OFF_YH     ((size_t)64 << 20)          // f16 y        [BT, C]      8 MB

// ---------------- CDNA5 async global->LDS staging ----------------
#if defined(__has_builtin)
#  if __has_builtin(__builtin_amdgcn_global_load_async_to_lds_b128)
#    define HAVE_ASYNC_LDS 1
#  endif
#endif
#ifndef HAVE_ASYNC_LDS
#  define HAVE_ASYNC_LDS 0
#endif

__device__ __forceinline__ void stage16(const _Float16* g, _Float16* l) {
#if HAVE_ASYNC_LDS
    // signature: (global v4i*, local v4i*, imm offset, imm cpol)
    __builtin_amdgcn_global_load_async_to_lds_b128(
        (__attribute__((address_space(1))) v4i*)(v4i*)g,
        (__attribute__((address_space(3))) v4i*)(v4i*)l, 0, 0);
#else
    *(v8h*)l = *(const v8h*)g;            // fallback: VGPR bounce + ds_store
#endif
}

template <int N>
__device__ __forceinline__ void wait_slab() {
#if HAVE_ASYNC_LDS
#  if __has_builtin(__builtin_amdgcn_s_wait_asynccnt)
    __builtin_amdgcn_s_wait_asynccnt(N);
#  else
    asm volatile("s_wait_asynccnt %0" :: "i"(N) : "memory");
#  endif
#endif
}

// ---------------- elementwise converters ----------------
__global__ void cvt_f32_to_f16(const float* __restrict__ in,
                               _Float16* __restrict__ out, int n) {
    int i = blockIdx.x * blockDim.x + threadIdx.x;
    if (i < n) out[i] = (_Float16)in[i];
}

// WT[n*K + k] = W[k*N + n]
__global__ void transpose_to_f16(const float* __restrict__ W,
                                 _Float16* __restrict__ WT, int K, int N) {
    int i = blockIdx.x * blockDim.x + threadIdx.x;
    if (i >= K * N) return;
    int n = i / K, k = i % K;
    WT[(size_t)n * K + k] = (_Float16)W[(size_t)k * N + n];
}

// ---------------- WMMA GEMM with LDS-staged weights ----------------
// C[M,N] = A[M,K] @ WT[N,K]^T + bias
// 128 threads = 4 waves per workgroup; workgroup tile = 64(M) x 64(N).
// Wave w owns rows [m0 + 16w, +16). W slab (64 n-rows x 32 k) is staged into
// double-buffered LDS with async global->LDS DMA and shared by all 4 waves.
template <bool OUT_F16>
__global__ __launch_bounds__(128)
void wmma_gemm(const _Float16* __restrict__ A, const _Float16* __restrict__ WT,
               const float* __restrict__ bias, void* __restrict__ Cout,
               int M, int N, int K) {
    __shared__ __align__(16) _Float16 Wlds[2][64 * 32];   // 2 x 4KB

    const int tid  = threadIdx.x;
    const int wave = tid >> 5;
    const int lane = tid & 31;
    const int r16  = lane & 15;
    const int half = lane >> 4;
    const int m0 = blockIdx.x * 64 + wave * 16;
    const int n0 = blockIdx.y * 64;

    const v8f zf = {0.f, 0.f, 0.f, 0.f, 0.f, 0.f, 0.f, 0.f};
    v8f acc[4] = {zf, zf, zf, zf};

    const _Float16* arow = A + (size_t)(m0 + r16) * K;
    const int ac0 = half ? 8  : 0;   // A chunk0 K-offset, chunk1 = +16
    const int bc0 = half ? 8  : 0;   // B chunk0 K-offset, chunk1 = +16

    // stage one 64x32 W slab (k..k+31 for n0..n0+63) into LDS buffer `buf`
    auto issue_slab = [&](int k, int buf) {
#pragma unroll
        for (int rep = 0; rep < 2; ++rep) {
            const int u   = tid + rep * 128;   // 256 units of 8 halves (16B)
            const int row = u >> 2;
            const int uu  = u & 3;
            stage16(WT + (size_t)(n0 + row) * K + k + uu * 8,
                    &Wlds[buf][row * 32 + uu * 8]);
        }
    };

    const int NK = K / 32;
    issue_slab(0, 0);
    for (int ki = 0; ki < NK; ++ki) {
        const int k = ki * 32;
        if (ki + 1 < NK) {                       // prefetch next slab (other buf)
            issue_slab(k + 32, (ki + 1) & 1);
            wait_slab<2>();                      // wait current slab only
        } else {
            wait_slab<0>();
        }
        __syncthreads();                         // slab visible to all waves

        V16 af;
        af.h[0] = *(const v8h*)(arow + k + ac0);
        af.h[1] = *(const v8h*)(arow + k + ac0 + 16);
        const _Float16* wb = &Wlds[ki & 1][0];
#pragma unroll
        for (int nt = 0; nt < 4; ++nt) {
            const _Float16* brow = wb + (nt * 16 + r16) * 32;
            V16 bf;
            bf.h[0] = *(const v8h*)(brow + bc0);
            bf.h[1] = *(const v8h*)(brow + bc0 + 16);
            acc[nt] = __builtin_amdgcn_wmma_f32_16x16x32_f16(
                false, af.v, false, bf.v, (short)0, acc[nt], false, false);
        }
        __syncthreads();                         // done reading before overwrite
    }
#pragma unroll
    for (int nt = 0; nt < 4; ++nt) {
        const int n = n0 + nt * 16 + r16;
        const float bv = bias[n];
#pragma unroll
        for (int j = 0; j < 8; ++j) {
            const int m = m0 + j + half * 8;
            const float v = acc[nt][j] + bv;
            if (OUT_F16) ((_Float16*)Cout)[(size_t)m * N + n] = (_Float16)v;
            else         ((float*)Cout)[(size_t)m * N + n]    = v;
        }
    }
}

// ---------------- RoPE + head split/transpose ----------------
// thread per (b, t, h, i) with i in [0, 32): rotates pair (2i, 2i+1)
__global__ void rope_split(const _Float16* __restrict__ qkv,
                           const float* __restrict__ cosT,
                           const float* __restrict__ sinT,
                           _Float16* __restrict__ Qh, _Float16* __restrict__ Kh,
                           _Float16* __restrict__ VT) {
    const int idx = blockIdx.x * blockDim.x + threadIdx.x;
    if (idx >= BATCH * SEQ * N_HEAD * 32) return;
    const int i = idx & 31;
    const int h = (idx >> 5) & 15;
    const int t = (idx >> 9) & (SEQ - 1);
    const int b = idx >> 20;

    const float c = cosT[t * 32 + i];
    const float s = sinT[t * 32 + i];

    const size_t row = ((size_t)b * SEQ + t) * C3;
    const int co = h * HEAD_DIM + 2 * i;
    const float q0 = (float)qkv[row + co],              q1 = (float)qkv[row + co + 1];
    const float k0 = (float)qkv[row + N_EMBD + co],     k1 = (float)qkv[row + N_EMBD + co + 1];
    const float v0 = (float)qkv[row + 2 * N_EMBD + co], v1 = (float)qkv[row + 2 * N_EMBD + co + 1];

    const int bh = b * N_HEAD + h;
    const size_t qkbase = (((size_t)bh * SEQ) + t) * HEAD_DIM + 2 * i;
    // fold softmax scale 1/sqrt(D) = 0.125 into q
    Qh[qkbase]     = (_Float16)((q0 * c - q1 * s) * 0.125f);
    Qh[qkbase + 1] = (_Float16)((q0 * s + q1 * c) * 0.125f);
    Kh[qkbase]     = (_Float16)(k0 * c - k1 * s);
    Kh[qkbase + 1] = (_Float16)(k0 * s + k1 * c);
    // v transposed: [BH, D, T]
    const size_t vb = ((size_t)bh * HEAD_DIM + 2 * i) * SEQ + t;
    VT[vb]       = (_Float16)v0;
    VT[vb + SEQ] = (_Float16)v1;
}

// ---------------- flash attention: one wave per 16-row q tile ----------------
__global__ __launch_bounds__(32)
void flash_attn(const _Float16* __restrict__ Qh, const _Float16* __restrict__ Kh,
                const _Float16* __restrict__ VT, _Float16* __restrict__ Y) {
    __shared__ __align__(32) _Float16 Pst[16 * 32];

    const int lane = threadIdx.x & 31;
    const int r16  = lane & 15;
    const int half = lane >> 4;
    const int m0 = blockIdx.x * 16;
    const int bh = blockIdx.y;
    const int b = bh >> 4, h = bh & 15;

    const _Float16* qb = Qh + (size_t)bh * SEQ * HEAD_DIM;
    const _Float16* kb = Kh + (size_t)bh * SEQ * HEAD_DIM;
    const _Float16* vb = VT + (size_t)bh * HEAD_DIM * SEQ;

    const int ac0 = half ? 8  : 0;
    const int bc0 = half ? 8  : 0;

    // Q fragments (D=64 -> two K=32 A-fragments), loaded once
    V16 qf[2];
    const _Float16* qrow = qb + (size_t)(m0 + r16) * HEAD_DIM;
#pragma unroll
    for (int kk = 0; kk < 2; ++kk) {
        qf[kk].h[0] = *(const v8h*)(qrow + kk * 32 + ac0);
        qf[kk].h[1] = *(const v8h*)(qrow + kk * 32 + ac0 + 16);
    }

    const v8f zf = {0.f, 0.f, 0.f, 0.f, 0.f, 0.f, 0.f, 0.f};
    v8f O[4] = {zf, zf, zf, zf};
    float mrow[8], lrow[8];
#pragma unroll
    for (int j = 0; j < 8; ++j) { mrow[j] = -1e30f; lrow[j] = 0.f; }

    for (int s0 = 0; s0 <= m0; s0 += 32) {
        // ---- S = q @ k^T : two 16x16 tiles over 32 keys ----
        v8f S[2] = {zf, zf};
#pragma unroll
        for (int tile = 0; tile < 2; ++tile) {
            int r = s0 + tile * 16 + r16;
            if (r > SEQ - 1) r = SEQ - 1;           // clamp (masked anyway)
            const _Float16* krow = kb + (size_t)r * HEAD_DIM;
#pragma unroll
            for (int kk = 0; kk < 2; ++kk) {
                V16 bf;
                bf.h[0] = *(const v8h*)(krow + kk * 32 + bc0);
                bf.h[1] = *(const v8h*)(krow + kk * 32 + bc0 + 16);
                S[tile] = __builtin_amdgcn_wmma_f32_16x16x32_f16(
                    false, qf[kk].v, false, bf.v, (short)0, S[tile], false, false);
            }
        }
        // ---- causal mask + online softmax (C-layout: row = j + 8*half, col = r16) ----
#pragma unroll
        for (int j = 0; j < 8; ++j) {
            const int trow = m0 + j + half * 8;
            const float s0v = (s0 + r16      <= trow) ? S[0][j] : -1e30f;
            const float s1v = (s0 + 16 + r16 <= trow) ? S[1][j] : -1e30f;
            float mx = fmaxf(s0v, s1v);
            mx = fmaxf(mx, __shfl_xor(mx, 1, 32));
            mx = fmaxf(mx, __shfl_xor(mx, 2, 32));
            mx = fmaxf(mx, __shfl_xor(mx, 4, 32));
            mx = fmaxf(mx, __shfl_xor(mx, 8, 32));
            const float mnew = fmaxf(mrow[j], mx);
            const float sc = __expf(mrow[j] - mnew);
            const float p0 = __expf(s0v - mnew);
            const float p1 = __expf(s1v - mnew);
            float rs = p0 + p1;
            rs += __shfl_xor(rs, 1, 32);
            rs += __shfl_xor(rs, 2, 32);
            rs += __shfl_xor(rs, 4, 32);
            rs += __shfl_xor(rs, 8, 32);
            lrow[j] = lrow[j] * sc + rs;
            mrow[j] = mnew;
#pragma unroll
            for (int nt = 0; nt < 4; ++nt) O[nt][j] *= sc;
            // stage P into LDS for C-layout -> A-fragment relayout
            const int mloc = j + half * 8;
            Pst[mloc * 32 + r16]      = (_Float16)p0;
            Pst[mloc * 32 + 16 + r16] = (_Float16)p1;
        }
        __syncthreads();
        // ---- P (16x32) as A-fragment ----
        V16 pf;
        pf.h[0] = *(const v8h*)(&Pst[r16 * 32 + ac0]);
        pf.h[1] = *(const v8h*)(&Pst[r16 * 32 + ac0 + 16]);
        // ---- O += P @ V : B-fragment rows come from v^T (contiguous keys) ----
#pragma unroll
        for (int nt = 0; nt < 4; ++nt) {
            const _Float16* vrow = vb + (size_t)(nt * 16 + r16) * SEQ + s0;
            V16 bf;
            bf.h[0] = *(const v8h*)(vrow + bc0);
            bf.h[1] = *(const v8h*)(vrow + bc0 + 16);
            O[nt] = __builtin_amdgcn_wmma_f32_16x16x32_f16(
                false, pf.v, false, bf.v, (short)0, O[nt], false, false);
        }
        __syncthreads();
    }
    // ---- normalize + write y in [B, T, C] layout ----
#pragma unroll
    for (int j = 0; j < 8; ++j) {
        const float inv = 1.0f / lrow[j];
        const int trow = m0 + j + half * 8;
        _Float16* yrow = Y + ((size_t)b * SEQ + trow) * N_EMBD + h * HEAD_DIM;
#pragma unroll
        for (int nt = 0; nt < 4; ++nt)
            yrow[nt * 16 + r16] = (_Float16)(O[nt][j] * inv);
    }
}

// ---------------- launcher ----------------
extern "C" void kernel_launch(void* const* d_in, const int* in_sizes, int n_in,
                              void* d_out, int out_size, void* d_ws, size_t ws_size,
                              hipStream_t stream) {
    const float* x     = (const float*)d_in[0];
    const float* Wqkv  = (const float*)d_in[1];
    const float* bqkv  = (const float*)d_in[2];
    const float* Wproj = (const float*)d_in[3];
    const float* bproj = (const float*)d_in[4];
    const float* cosT  = (const float*)d_in[5];
    const float* sinT  = (const float*)d_in[6];

    char* ws = (char*)d_ws;
    _Float16* xh     = (_Float16*)(ws + OFF_XH);
    _Float16* WqkvT  = (_Float16*)(ws + OFF_WQKVT);
    _Float16* WprojT = (_Float16*)(ws + OFF_WPROJT);
    _Float16* qkv    = (_Float16*)(ws + OFF_QKV);
    _Float16* Qh     = (_Float16*)(ws + OFF_QH);
    _Float16* Kh     = (_Float16*)(ws + OFF_KH);
    _Float16* VT     = (_Float16*)(ws + OFF_VT);
    _Float16* Yh     = (_Float16*)(ws + OFF_YH);

    // 1) convert x to f16
    cvt_f32_to_f16<<<(ROWS_BT * N_EMBD + 255) / 256, 256, 0, stream>>>(x, xh, ROWS_BT * N_EMBD);
    // 2) transpose weights to [N, K] f16
    transpose_to_f16<<<(N_EMBD * C3 + 255) / 256, 256, 0, stream>>>(Wqkv, WqkvT, N_EMBD, C3);
    transpose_to_f16<<<(N_EMBD * N_EMBD + 255) / 256, 256, 0, stream>>>(Wproj, WprojT, N_EMBD, N_EMBD);
    // 3) qkv = x @ Wqkv + bqkv   (M=4096, N=3072, K=1024)
    wmma_gemm<true><<<dim3(ROWS_BT / 64, C3 / 64), 128, 0, stream>>>(
        xh, WqkvT, bqkv, qkv, ROWS_BT, C3, N_EMBD);
    // 4) RoPE + head split + v transpose
    rope_split<<<(BATCH * SEQ * N_HEAD * 32 + 255) / 256, 256, 0, stream>>>(
        qkv, cosT, sinT, Qh, Kh, VT);
    // 5) causal flash attention
    flash_attn<<<dim3(SEQ / 16, BATCH * N_HEAD), 32, 0, stream>>>(Qh, Kh, VT, Yh);
    // 6) out = y @ Wproj + bproj  (M=4096, N=1024, K=1024), f32 output
    wmma_gemm<false><<<dim3(ROWS_BT / 64, N_EMBD / 64), 128, 0, stream>>>(
        Yh, WprojT, bproj, d_out, ROWS_BT, N_EMBD, N_EMBD);
}
